// ProposalNet_59416577573159
// MI455X (gfx1250) — matrix operand order repaired and checked by reading, not compiled, not run on hardware
//
#include <hip/hip_runtime.h>
#include <hip/hip_bf16.h>

#define BB 32
#define NN 2048
#define CC 256
#define SS 128
#define NSAMPLE 16
#define RADIUS_F 0.3f
#define NHEAD 12
#define KPAD0 288            // 259 padded to multiple of 32
#define ROWS (BB * SS * NSAMPLE)   // 65536
#define PROWS (BB * SS)            // 4096
#define OUTD 124
#define EPSV 1e-5f

typedef __attribute__((ext_vector_type(16))) _Float16 v16h;
typedef __attribute__((ext_vector_type(8)))  _Float16 v8h;
typedef __attribute__((ext_vector_type(8)))  float    v8f;

__constant__ float MEAN_SIZES_C[30] = {
    0.76584f, 1.398258f, 0.472728f,
    2.114256f, 1.6203f,   0.927272f,
    0.404671f, 1.071108f, 1.688889f,
    0.591958f, 0.552978f, 0.827272f,
    0.69519f,  1.346299f, 0.736364f,
    0.528526f, 1.002642f, 1.172878f,
    0.500618f, 0.632163f, 0.683424f,
    0.923508f, 1.867419f, 0.845495f,
    0.791118f, 1.279516f, 0.718182f,
    0.699104f, 0.454178f, 0.75625f
};

// ---------------------------------------------------------------------------
// 1) Farthest-point sampling: one workgroup per batch, xyz + mindist in LDS.
// ---------------------------------------------------------------------------
__global__ void fps_kernel(const float* __restrict__ xyz,
                           int* __restrict__ fpsIdx,
                           float* __restrict__ newXyz) {
    const int b = blockIdx.x;
    const int t = threadIdx.x;
    __shared__ float px[NN], py[NN], pz[NN], mind[NN];
    __shared__ float rv[256];
    __shared__ int   ri[256];

    for (int j = t; j < NN; j += 256) {
        px[j]   = xyz[((long)b * NN + j) * 3 + 0];
        py[j]   = xyz[((long)b * NN + j) * 3 + 1];
        pz[j]   = xyz[((long)b * NN + j) * 3 + 2];
        mind[j] = 1e10f;
    }
    __syncthreads();

    int last = 0;
    for (int s = 0; s < SS; ++s) {
        if (t == 0) {
            fpsIdx[b * SS + s] = last;
            newXyz[(b * SS + s) * 3 + 0] = px[last];
            newXyz[(b * SS + s) * 3 + 1] = py[last];
            newXyz[(b * SS + s) * 3 + 2] = pz[last];
        }
        const float cx = px[last], cy = py[last], cz = pz[last];
        float bv = -1.0f;
        int   bi = 0;
        for (int j = t; j < NN; j += 256) {
            float dx = px[j] - cx, dy = py[j] - cy, dz = pz[j] - cz;
            float d  = dx * dx + dy * dy + dz * dz;
            float md = fminf(mind[j], d);
            mind[j] = md;
            if (md > bv) { bv = md; bi = j; }
        }
        rv[t] = bv; ri[t] = bi;
        __syncthreads();
        for (int off = 128; off > 0; off >>= 1) {
            if (t < off) {
                float ov = rv[t + off]; int oi = ri[t + off];
                if (ov > rv[t] || (ov == rv[t] && oi < ri[t])) { rv[t] = ov; ri[t] = oi; }
            }
            __syncthreads();
        }
        last = ri[0];
        __syncthreads();
    }
}

// ---------------------------------------------------------------------------
// 2) Ball query: first NSAMPLE indices (ascending) within radius.
// ---------------------------------------------------------------------------
__global__ void ballq_kernel(const float* __restrict__ xyz,
                             const float* __restrict__ newXyz,
                             int* __restrict__ ballIdx) {
    int q = blockIdx.x * blockDim.x + threadIdx.x;
    if (q >= PROWS) return;
    int b = q / SS;
    float cx = newXyz[q * 3 + 0];
    float cy = newXyz[q * 3 + 1];
    float cz = newXyz[q * 3 + 2];
    const float* base = xyz + (long)b * NN * 3;
    const float r2 = RADIUS_F * RADIUS_F;
    int idxs[NSAMPLE];
    int cnt = 0;
    for (int j = 0; j < NN && cnt < NSAMPLE; ++j) {
        float dx = base[j * 3 + 0] - cx;
        float dy = base[j * 3 + 1] - cy;
        float dz = base[j * 3 + 2] - cz;
        if (dx * dx + dy * dy + dz * dz < r2) idxs[cnt++] = j;
    }
    int f = (cnt > 0) ? idxs[0] : 0;
    for (int k = cnt; k < NSAMPLE; ++k) idxs[k] = f;
    for (int k = 0; k < NSAMPLE; ++k) ballIdx[q * NSAMPLE + k] = idxs[k];
}

// ---------------------------------------------------------------------------
// 3) Gather grouped input (B,S,NSAMP, 3+C) -> f16 matrix [ROWS x KPAD0].
// ---------------------------------------------------------------------------
__global__ void gather_kernel(const float* __restrict__ xyz,
                              const float* __restrict__ feat,
                              const float* __restrict__ newXyz,
                              const int* __restrict__ ballIdx,
                              _Float16* __restrict__ Xg) {
    int gid = blockIdx.x * blockDim.x + threadIdx.x;
    const int total = ROWS * KPAD0;
    if (gid >= total) return;
    int row = gid / KPAD0;
    int c   = gid % KPAD0;
    int q   = row / NSAMPLE;            // (b,s) flat
    int b   = q / SS;
    int j   = ballIdx[row];
    float val = 0.0f;
    if (c < 3) {
        val = (xyz[((long)b * NN + j) * 3 + c] - newXyz[q * 3 + c]) * (1.0f / RADIUS_F);
    } else if (c < 3 + CC) {
        val = feat[((long)b * CC + (c - 3)) * NN + j];
    }
    Xg[gid] = (_Float16)val;
}

// ---------------------------------------------------------------------------
// 4) Fold BN into weights/bias, cast to f16, and PACK into WMMA B-fragment
//    order: Bp[((nTile*kblocks + kb)*32 + lane)*16 + i] = B[kb*32 + 16*(i/8) + 8*hi + i%8]
//    [nTile*16 + (lane&15)] so each lane's 16 halves are one 32B chunk.
// ---------------------------------------------------------------------------
__global__ void prep_pack_kernel(const float* __restrict__ w, const float* __restrict__ bb,
                                 const float* __restrict__ g, const float* __restrict__ be,
                                 const float* __restrict__ m, const float* __restrict__ v,
                                 _Float16* __restrict__ Bp, float* __restrict__ biasOut,
                                 int Kin, int Kpad, int Nin, int Npad, int hasBN) {
    int gid = blockIdx.x * blockDim.x + threadIdx.x;
    if (gid < Npad) {
        float bias = 0.0f;
        if (gid < Nin) {
            if (hasBN) {
                float s = g[gid] / sqrtf(v[gid] + EPSV);
                bias = (bb[gid] - m[gid]) * s + be[gid];
            } else {
                bias = bb[gid];
            }
        }
        biasOut[gid] = bias;
    }
    if (gid >= Kpad * Npad) return;
    const int kblocks = Kpad >> 5;
    int i    = gid & 15;
    int lane = (gid >> 4) & 31;
    int rest = gid >> 9;                 // nTile*kblocks + kb
    int kb    = rest % kblocks;
    int nTile = rest / kblocks;
    int n = nTile * 16 + (lane & 15);
    int k = kb * 32 + ((i >> 3) << 4) + ((lane >> 4) << 3) + (i & 7);
    float val = 0.0f;
    if (k < Kin && n < Nin) {
        float s = hasBN ? (g[n] / sqrtf(v[n] + EPSV)) : 1.0f;
        val = w[k * Nin + n] * s;
    }
    Bp[gid] = (_Float16)val;
}

// ---------------------------------------------------------------------------
// 5) WMMA GEMM: one wave computes a 16 x (16*NT) output strip.
//    Packed weights staged in LDS once per block; B fragments via ds_load_b128.
//    A fragments double-buffered in registers (global b128 loads).
// ---------------------------------------------------------------------------
union AFrag { v16h v; v8h h[2]; _Float16 e[16]; };
union CFrag { v8f  v; float    e[8];  };

template <int NT, int KBLOCKS, int RELU, int OUTHALF>
__launch_bounds__(256)
__global__ void wmma_gemm_kernel(const _Float16* __restrict__ A,
                                 const _Float16* __restrict__ Bp,
                                 const float* __restrict__ bias,
                                 void* __restrict__ outP,
                                 int M, int Ncols) {
    constexpr int K       = KBLOCKS * 32;
    constexpr int LDSHALF = NT * KBLOCKS * 512;   // packed weights, halves
    __shared__ alignas(32) _Float16 ldsB[LDSHALF];

    const int t = threadIdx.x;
    // cooperative stage: packed weights global -> LDS (b128 both sides)
    for (int i = t * 8; i < LDSHALF; i += 256 * 8)
        *(v8h*)(ldsB + i) = *(const v8h*)(Bp + i);
    __syncthreads();

    const int wave = (blockIdx.x * blockDim.x + t) >> 5;
    const int lane = t & 31;
    if (wave * 16 >= M) return;                 // wave-uniform; after barrier

    const int lo = lane & 15;
    const int hi = lane >> 4;
    const _Float16* arow = A + (long)(wave * 16 + lo) * K;
    const _Float16* lrow = ldsB + lane * 16;

    CFrag c[NT];
#pragma unroll
    for (int nt = 0; nt < NT; ++nt)
#pragma unroll
        for (int r = 0; r < 8; ++r) c[nt].e[r] = 0.0f;

    AFrag aCur, aNxt;
    aCur.h[0] = *(const v8h*)(arow + 8 * hi);
    aCur.h[1] = *(const v8h*)(arow + 16 + 8 * hi);

    for (int kb = 0; kb < KBLOCKS; ++kb) {
        if (kb + 1 < KBLOCKS) {                 // prefetch next A fragment
            aNxt.h[0] = *(const v8h*)(arow + (kb + 1) * 32 + 8 * hi);
            aNxt.h[1] = *(const v8h*)(arow + (kb + 1) * 32 + 16 + 8 * hi);
        }
        AFrag bf[NT];                           // batch all B frags from LDS
#pragma unroll
        for (int nt = 0; nt < NT; ++nt)
            bf[nt].v = *(const v16h*)(lrow + (nt * KBLOCKS + kb) * 512);
#pragma unroll
        for (int nt = 0; nt < NT; ++nt)
            c[nt].v = __builtin_amdgcn_wmma_f32_16x16x32_f16(
                false, aCur.v, false, bf[nt].v, (short)0, c[nt].v, false, false);
        aCur = aNxt;
    }

#pragma unroll
    for (int nt = 0; nt < NT; ++nt) {
        const int ncol = nt * 16 + lo;
        const float bval = bias[ncol];
#pragma unroll
        for (int r = 0; r < 8; ++r) {
            long row = (long)wave * 16 + r + 8 * hi;
            float val = c[nt].e[r] + bval;
            if (RELU) val = fmaxf(val, 0.0f);
            long off = row * Ncols + ncol;
            if (OUTHALF) ((_Float16*)outP)[off] = (_Float16)val;
            else         ((float*)outP)[off]    = val;
        }
    }
}

// ---------------------------------------------------------------------------
// 6) Max over NSAMPLE grouped points.
// ---------------------------------------------------------------------------
__global__ void maxpool_kernel(const _Float16* __restrict__ H,
                               _Float16* __restrict__ P) {
    int gid = blockIdx.x * blockDim.x + threadIdx.x;
    if (gid >= PROWS * 128) return;
    int c = gid % 128;
    int r = gid / 128;
    float mx = -1e30f;
    for (int k = 0; k < NSAMPLE; ++k)
        mx = fmaxf(mx, (float)H[((long)r * NSAMPLE + k) * 128 + c]);
    P[gid] = (_Float16)mx;
}

// ---------------------------------------------------------------------------
// 7) Assemble 124-d output from net[PROWS x 80] + new_xyz.
// ---------------------------------------------------------------------------
__global__ void assemble_kernel(const float* __restrict__ net,
                                const float* __restrict__ newXyz,
                                float* __restrict__ out) {
    int gid = blockIdx.x * blockDim.x + threadIdx.x;
    if (gid >= PROWS * OUTD) return;
    int j = gid % OUTD;
    int r = gid / OUTD;
    const float* nr = net + (long)r * 80;
    float val;
    if (j < 3)        val = newXyz[r * 3 + j];
    else if (j < 5)   val = nr[j - 3];                              // obj
    else if (j < 8)   val = newXyz[r * 3 + (j - 5)] + nr[2 + (j - 5)]; // ctr
    else if (j < 20)  val = nr[5  + (j - 8)];                       // hs
    else if (j < 32)  val = nr[17 + (j - 20)];                      // hrn
    else if (j < 44)  val = nr[17 + (j - 32)] * (3.14159265358979f / NHEAD); // hr
    else if (j < 54)  val = nr[29 + (j - 44)];                      // ss
    else if (j < 84)  val = nr[39 + (j - 54)];                      // srn
    else if (j < 114) { int jj = j - 84; val = nr[39 + jj] * MEAN_SIZES_C[jj]; } // sr
    else              val = nr[69 + (j - 114)];                     // sem
    out[gid] = val;
}

// ---------------------------------------------------------------------------
extern "C" void kernel_launch(void* const* d_in, const int* in_sizes, int n_in,
                              void* d_out, int out_size, void* d_ws, size_t ws_size,
                              hipStream_t stream) {
    const float* xyz  = (const float*)d_in[0];
    const float* feat = (const float*)d_in[1];
    const float* w0  = (const float*)d_in[2];
    const float* bb0 = (const float*)d_in[3];
    const float* g0  = (const float*)d_in[4];
    const float* be0 = (const float*)d_in[5];
    const float* m0  = (const float*)d_in[6];
    const float* v0  = (const float*)d_in[7];
    const float* w1  = (const float*)d_in[8];
    const float* bb1 = (const float*)d_in[9];
    const float* g1  = (const float*)d_in[10];
    const float* be1 = (const float*)d_in[11];
    const float* m1  = (const float*)d_in[12];
    const float* v1  = (const float*)d_in[13];
    const float* w2  = (const float*)d_in[14];
    const float* bb2 = (const float*)d_in[15];
    const float* g2  = (const float*)d_in[16];
    const float* be2 = (const float*)d_in[17];
    const float* m2  = (const float*)d_in[18];
    const float* v2  = (const float*)d_in[19];
    const float* c1w = (const float*)d_in[20];
    const float* c1b = (const float*)d_in[21];
    const float* g3  = (const float*)d_in[22];
    const float* be3 = (const float*)d_in[23];
    const float* m3  = (const float*)d_in[24];
    const float* v3  = (const float*)d_in[25];
    const float* c2w = (const float*)d_in[26];
    const float* c2b = (const float*)d_in[27];
    const float* g4  = (const float*)d_in[28];
    const float* be4 = (const float*)d_in[29];
    const float* m4  = (const float*)d_in[30];
    const float* v4  = (const float*)d_in[31];
    const float* c3w = (const float*)d_in[32];
    const float* c3b = (const float*)d_in[33];

    // --- workspace carve-up (256B aligned) ---
    size_t off = 0;
    auto wsAlloc = [&](size_t bytes) -> void* {
        void* p = (char*)d_ws + off;
        off += (bytes + 255) & ~(size_t)255;
        return p;
    };
    int*      fpsIdx  = (int*)      wsAlloc((size_t)PROWS * 4);
    float*    newXyz  = (float*)    wsAlloc((size_t)PROWS * 3 * 4);
    int*      ballIdx = (int*)      wsAlloc((size_t)ROWS * 4);
    _Float16* Xg      = (_Float16*) wsAlloc((size_t)ROWS * KPAD0 * 2);
    _Float16* H0      = (_Float16*) wsAlloc((size_t)ROWS * 128 * 2);
    _Float16* H1      = (_Float16*) wsAlloc((size_t)ROWS * 128 * 2);
    _Float16* P       = (_Float16*) wsAlloc((size_t)PROWS * 128 * 2);
    _Float16* Q       = (_Float16*) wsAlloc((size_t)PROWS * 128 * 2);
    _Float16* R       = (_Float16*) wsAlloc((size_t)PROWS * 128 * 2);
    float*    net     = (float*)    wsAlloc((size_t)PROWS * 80 * 4);
    _Float16* W0p = (_Float16*) wsAlloc((size_t)KPAD0 * 128 * 2);
    _Float16* W1p = (_Float16*) wsAlloc((size_t)128 * 128 * 2);
    _Float16* W2p = (_Float16*) wsAlloc((size_t)128 * 128 * 2);
    _Float16* C1p = (_Float16*) wsAlloc((size_t)128 * 128 * 2);
    _Float16* C2p = (_Float16*) wsAlloc((size_t)128 * 128 * 2);
    _Float16* C3p = (_Float16*) wsAlloc((size_t)128 * 80 * 2);
    float* b0 = (float*) wsAlloc(128 * 4);
    float* b1 = (float*) wsAlloc(128 * 4);
    float* b2 = (float*) wsAlloc(128 * 4);
    float* b3 = (float*) wsAlloc(128 * 4);
    float* b4 = (float*) wsAlloc(128 * 4);
    float* b5 = (float*) wsAlloc(80 * 4);
    (void)ws_size; (void)n_in; (void)in_sizes; (void)out_size; (void)fpsIdx;

    // 1) FPS
    fps_kernel<<<BB, 256, 0, stream>>>(xyz, fpsIdx, newXyz);
    // 2) ball query
    ballq_kernel<<<(PROWS + 255) / 256, 256, 0, stream>>>(xyz, newXyz, ballIdx);
    // 3) gather -> f16
    {
        int total = ROWS * KPAD0;
        gather_kernel<<<(total + 255) / 256, 256, 0, stream>>>(xyz, feat, newXyz, ballIdx, Xg);
    }
    // 4) weight prep (BN folded + fragment packing)
    {
        int t0 = KPAD0 * 128, t1 = 128 * 128, t5 = 128 * 80;
        prep_pack_kernel<<<(t0 + 255) / 256, 256, 0, stream>>>(w0,  bb0, g0, be0, m0, v0, W0p, b0, 259, KPAD0, 128, 128, 1);
        prep_pack_kernel<<<(t1 + 255) / 256, 256, 0, stream>>>(w1,  bb1, g1, be1, m1, v1, W1p, b1, 128, 128, 128, 128, 1);
        prep_pack_kernel<<<(t1 + 255) / 256, 256, 0, stream>>>(w2,  bb2, g2, be2, m2, v2, W2p, b2, 128, 128, 128, 128, 1);
        prep_pack_kernel<<<(t1 + 255) / 256, 256, 0, stream>>>(c1w, c1b, g3, be3, m3, v3, C1p, b3, 128, 128, 128, 128, 1);
        prep_pack_kernel<<<(t1 + 255) / 256, 256, 0, stream>>>(c2w, c2b, g4, be4, m4, v4, C2p, b4, 128, 128, 128, 128, 1);
        prep_pack_kernel<<<(t5 + 255) / 256, 256, 0, stream>>>(c3w, c3b, nullptr, nullptr, nullptr, nullptr, C3p, b5, 128, 128, 79, 80, 0);
    }
    // 5) WMMA GEMM chain (one wave per 16-row strip, all N tiles per wave)
    {
        // big layers: M=65536 -> 4096 waves -> 512 blocks of 256
        wmma_gemm_kernel<8, 9, 1, 1><<<(ROWS / 16) * 32 / 256, 256, 0, stream>>>(
            Xg, W0p, b0, H0, ROWS, 128);
        wmma_gemm_kernel<8, 4, 1, 1><<<(ROWS / 16) * 32 / 256, 256, 0, stream>>>(
            H0, W1p, b1, H1, ROWS, 128);
        wmma_gemm_kernel<8, 4, 1, 1><<<(ROWS / 16) * 32 / 256, 256, 0, stream>>>(
            H1, W2p, b2, H0, ROWS, 128);
        maxpool_kernel<<<(PROWS * 128 + 255) / 256, 256, 0, stream>>>(H0, P);
        // small layers: M=4096 -> 256 waves -> 32 blocks
        wmma_gemm_kernel<8, 4, 1, 1><<<(PROWS / 16) * 32 / 256, 256, 0, stream>>>(
            P, C1p, b3, Q, PROWS, 128);
        wmma_gemm_kernel<8, 4, 1, 1><<<(PROWS / 16) * 32 / 256, 256, 0, stream>>>(
            Q, C2p, b4, R, PROWS, 128);
        wmma_gemm_kernel<5, 4, 0, 0><<<(PROWS / 16) * 32 / 256, 256, 0, stream>>>(
            R, C3p, b5, net, PROWS, 80);
    }
    // 7) output assembly
    assemble_kernel<<<(PROWS * OUTD + 255) / 256, 256, 0, stream>>>(net, newXyz, (float*)d_out);
}
